// Painn_60215441490041
// MI455X (gfx1250) — compile-verified, matrix operand order checked by reading, not run on hardware
//
#include <hip/hip_runtime.h>
#include <hip/hip_bf16.h>

#define NATOMS 20000
#define NEDGES 320000
#define FEAT   128
#define NRBF   20
#define NCONV  3
#define CUT    5.0f
#define AFEA   64
#define HFEA   128
#define NOUT   5

typedef _Float16 v16h __attribute__((ext_vector_type(16)));
typedef _Float16 v8h  __attribute__((ext_vector_type(8)));
typedef float    v8f  __attribute__((ext_vector_type(8)));

__device__ __forceinline__ float silu_fast(float x)
{
    // x * rcp(1+e^-x): v_exp_f32 + v_rcp_f32 instead of the IEEE div chain
    return x * __builtin_amdgcn_rcpf(1.0f + __expf(-x));
}

// ---------------------------------------------------------------------------
// WMMA GEMM, templated on K (64/128/256 -> K-loop fully unrolls).
// Block = 128 threads = 4 waves; each wave computes a 16x64 C strip
// (4 accumulators) so one A fragment feeds 4 WMMAs -> 4x less A traffic
// than a 16x16-per-wave kernel. Block covers a 64x64 C tile.
// The Kx64 f16 weight slice is staged once per block into LDS, transposed to
// [n][k] with the row padded to K+8 halves: row stride = (K+8)*2 bytes, NOT a
// multiple of the 256B bank period, so the per-lane ds_load_b128 fetches and
// the staging stores spread across banks (the unpadded K*2 stride aliases all
// 16 lanes onto one bank column).
// A: f32 with arbitrary strides (covers the nfd,fg->ngd einsum with ac=3),
// converted to f16 in-register; accumulation in f32.
// Lane layout per CDNA5 ISA 7.12.2:
//   lanes 0-15 : row/col = lane,    K halves {kk+0..7, kk+16..23}
//   lanes 16-31: row/col = lane-16, K halves {kk+8..15, kk+24..31}
// D layout: VGPR r -> (row = half*8 + r, col).
// ---------------------------------------------------------------------------
template <int K>
__global__ __launch_bounds__(128)
void gemm64x64_wmma(const float* __restrict__ A, long ar, long ac,
                    const _Float16* __restrict__ Bw, int N,
                    const float* __restrict__ bias,
                    float* __restrict__ C, long crow, long ccol,
                    int act, int Mtiles)
{
    constexpr int KP = K + 8;             // padded row: keeps 16B align, kills conflicts
    __shared__ _Float16 Bs[64 * KP];

    const int tid  = threadIdx.x;
    const int wave = tid >> 5;
    const int lane = tid & 31;
    const int half = lane >> 4;
    const int l16  = lane & 15;
    const int nbase = (int)blockIdx.x * 64;
    const int mtile = (int)blockIdx.y * 4 + wave;

    // ---- cooperative stage of B tile (K x 64) into LDS, transposed --------
    for (int it = 0; it < (K * 64) / 128; ++it) {
        const int idx = it * 128 + tid;
        const int k = idx >> 6;
        const int n = idx & 63;
        const int ncl = (nbase + n) < N ? (nbase + n) : (N - 1);
        Bs[n * KP + k] = Bw[(long)k * N + ncl];
    }
    __syncthreads();

    if (mtile < Mtiles) {                          // wave-uniform: EXEC stays all-1
        const long mrow = (long)mtile * 16 + l16;  // A row this lane feeds

        v8f acc[4];
#pragma unroll
        for (int t = 0; t < 4; ++t) {
            const int ncol = nbase + t * 16 + l16;
            const int ncl  = ncol < N ? ncol : (N - 1);
            const float bv = bias ? bias[ncl] : 0.0f;
#pragma unroll
            for (int r = 0; r < 8; ++r) acc[t][r] = bv;
        }

        const float* Ap = A + mrow * ar;
#pragma unroll
        for (int kk = 0; kk < K; kk += 32) {
            const int kb = kk + half * 8;
            v16h a;
#pragma unroll
            for (int j = 0; j < 8; ++j) {
                a[j]     = (_Float16)Ap[(long)(kb + j) * ac];
                a[j + 8] = (_Float16)Ap[(long)(kb + 16 + j) * ac];
            }
#pragma unroll
            for (int t = 0; t < 4; ++t) {
                const _Float16* Bl = &Bs[(t * 16 + l16) * KP];
                const v8h blo = *(const v8h*)(Bl + kb);        // ds_load_b128
                const v8h bhi = *(const v8h*)(Bl + kb + 16);   // ds_load_b128
                v16h b;
#pragma unroll
                for (int j = 0; j < 8; ++j) { b[j] = blo[j]; b[j + 8] = bhi[j]; }
                acc[t] = __builtin_amdgcn_wmma_f32_16x16x32_f16(
                    false, a, false, b, (short)0, acc[t], false, false);
            }
        }

        const long rbase = (long)mtile * 16 + half * 8;
#pragma unroll
        for (int t = 0; t < 4; ++t) {
            const int ncol = nbase + t * 16 + l16;
            if (ncol < N) {
#pragma unroll
                for (int r = 0; r < 8; ++r) {
                    float x = acc[t][r];
                    if (act) x = silu_fast(x);
                    C[(rbase + r) * crow + (long)ncol * ccol] = x;
                }
            }
        }
    }
}

// ---------------------------------------------------------------------------
// Small utility kernels
// ---------------------------------------------------------------------------
__global__ void cvt_f16_kernel(const float* __restrict__ in,
                               _Float16* __restrict__ out, long n)
{
    long i = (long)blockIdx.x * blockDim.x + threadIdx.x;
    if (i < n) out[i] = (_Float16)in[i];
}

__global__ void fill0_kernel(float* __restrict__ p, long n)
{
    long i = (long)blockIdx.x * blockDim.x + threadIdx.x;
    if (i < n) p[i] = 0.0f;
}

__global__ void addip_kernel(float* __restrict__ dst,
                             const float* __restrict__ src, long n)
{
    long i = (long)blockIdx.x * blockDim.x + threadIdx.x;
    if (i < n) dst[i] += src[i];
}

__global__ void embed_gather_kernel(const float* __restrict__ embed,
                                    const int* __restrict__ z,
                                    float* __restrict__ s)
{
    long i = (long)blockIdx.x * blockDim.x + threadIdx.x;
    if (i >= (long)NATOMS * FEAT) return;
    long n = i / FEAT, f = i % FEAT;
    s[i] = embed[(long)z[n] * FEAT + f];
}

// Per-edge geometry: unit vector, f_cut, 20 RBFs -> geo[e][24]
__global__ void edge_geo_kernel(const float* __restrict__ xyz,
                                const int* __restrict__ nbrs,
                                float* __restrict__ geo)
{
    int e = blockIdx.x * blockDim.x + threadIdx.x;
    if (e >= NEDGES) return;
    const int i = nbrs[2 * e + 0];
    const int j = nbrs[2 * e + 1];
    const float rx = xyz[3 * j + 0] - xyz[3 * i + 0];
    const float ry = xyz[3 * j + 1] - xyz[3 * i + 1];
    const float rz = xyz[3 * j + 2] - xyz[3 * i + 2];
    const float d  = sqrtf(rx * rx + ry * ry + rz * rz + 1e-12f);
    const float id = 1.0f / d;
    float* g = geo + (long)e * 24;
    g[0] = rx * id; g[1] = ry * id; g[2] = rz * id;
    const float pid = 3.14159265358979323846f * d / CUT;
    g[3] = (d < CUT) ? 0.5f * (cosf(pid) + 1.0f) : 0.0f;
#pragma unroll
    for (int r = 0; r < NRBF; ++r)
        g[4 + r] = sinf((float)(r + 1) * pid) * id;
}

// One block per edge, 128 threads = one feature each. Gathers phi[j], v[j],
// builds filter weights from the K=20 RBF projection in-register, and
// scatter-adds messages into ds/dv with f32 global atomics.
__global__ __launch_bounds__(128)
void edge_msg_kernel(const float* __restrict__ geo,
                     const int* __restrict__ nbrs,
                     const float* __restrict__ phi,
                     const float* __restrict__ v,
                     const float* __restrict__ Wd,   // [20][384] (layer slice)
                     const float* __restrict__ bd,   // [384]
                     float* __restrict__ ds,
                     float* __restrict__ dv)
{
    __shared__ float g[24];
    const int e = blockIdx.x;
    const int f = threadIdx.x;
    if (f < 24) g[f] = geo[(long)e * 24 + f];
    __syncthreads();

    const int i = nbrs[2 * e + 0];
    const int j = nbrs[2 * e + 1];
    const float fcut = g[3];

    float invc[3];
#pragma unroll
    for (int c = 0; c < 3; ++c) {
        const int col = c * FEAT + f;
        float w = bd[col];
#pragma unroll
        for (int r = 0; r < NRBF; ++r) w += g[4 + r] * Wd[r * (3 * FEAT) + col];
        invc[c] = phi[(long)j * (3 * FEAT) + col] * (w * fcut);
    }

    atomicAdd(&ds[(long)i * FEAT + f], invc[1]);
#pragma unroll
    for (int d2 = 0; d2 < 3; ++d2) {
        const float dvv = g[d2] * invc[2] + invc[0] * v[(long)j * (FEAT * 3) + f * 3 + d2];
        atomicAdd(&dv[(long)i * (FEAT * 3) + f * 3 + d2], dvv);
    }
}

// mix[n][0:128]=s, mix[n][128:256]=||v_v[n,f,:]||
__global__ void mix_kernel(const float* __restrict__ s,
                           const float* __restrict__ vv,
                           float* __restrict__ mix)
{
    long i = (long)blockIdx.x * blockDim.x + threadIdx.x;
    if (i >= (long)NATOMS * FEAT) return;
    long n = i / FEAT, f = i % FEAT;
    float sum = 1e-12f;
#pragma unroll
    for (int d = 0; d < 3; ++d) {
        float x = vv[n * (FEAT * 3) + f * 3 + d];
        sum += x * x;
    }
    mix[n * (2 * FEAT) + f]        = s[i];
    mix[n * (2 * FEAT) + FEAT + f] = sqrtf(sum);
}

// Gated update: v += u_v * a_vv ; s += <u_v,v_v> * a_sv + a_ss
__global__ void update_kernel(float* __restrict__ s, float* __restrict__ v,
                              const float* __restrict__ uv,
                              const float* __restrict__ vv,
                              const float* __restrict__ split)
{
    long i = (long)blockIdx.x * blockDim.x + threadIdx.x;
    if (i >= (long)NATOMS * FEAT) return;
    long n = i / FEAT, f = i % FEAT;
    const float a_vv = split[n * (3 * FEAT) + f];
    const float a_sv = split[n * (3 * FEAT) + FEAT + f];
    const float a_ss = split[n * (3 * FEAT) + 2 * FEAT + f];
    float dot = 0.0f;
#pragma unroll
    for (int d = 0; d < 3; ++d) {
        const long idx = n * (FEAT * 3) + f * 3 + d;
        const float u = uv[idx];
        dot += u * vv[idx];
        v[idx] += u * a_vv;
    }
    s[i] += dot * a_sv + a_ss;
}

// ---------------------------------------------------------------------------
// Host orchestration
// ---------------------------------------------------------------------------
static void* wsalloc(char* base, size_t* off, size_t bytes)
{
    void* p = base + *off;
    *off += (bytes + 255) & ~(size_t)255;
    return p;
}

extern "C" void kernel_launch(void* const* d_in, const int* in_sizes, int n_in,
                              void* d_out, int out_size, void* d_ws, size_t ws_size,
                              hipStream_t stream)
{
    (void)in_sizes; (void)n_in; (void)out_size; (void)ws_size;
    const float* xyz  = (const float*)d_in[0];
    const int*   z    = (const int*)d_in[1];
    const int*   nbrs = (const int*)d_in[2];
    const float* embd = (const float*)d_in[3];
    const float* W1 = (const float*)d_in[4],  *b1  = (const float*)d_in[5];
    const float* W2 = (const float*)d_in[6],  *b2  = (const float*)d_in[7];
    const float* Wd = (const float*)d_in[8],  *bd  = (const float*)d_in[9];
    const float* U  = (const float*)d_in[10], *V   = (const float*)d_in[11];
    const float* Ws1= (const float*)d_in[12], *bs1 = (const float*)d_in[13];
    const float* Ws2= (const float*)d_in[14], *bs2 = (const float*)d_in[15];
    const float* Wr1= (const float*)d_in[16], *br1 = (const float*)d_in[17];
    const float* Wr2= (const float*)d_in[18], *br2 = (const float*)d_in[19];
    const float* Wf1= (const float*)d_in[20], *bf1 = (const float*)d_in[21];
    const float* Wf2= (const float*)d_in[22], *bf2 = (const float*)d_in[23];
    const float* Wf3= (const float*)d_in[24], *bf3 = (const float*)d_in[25];
    float* out = (float*)d_out;

    char* base = (char*)d_ws;
    size_t off = 0;
    float* s      = (float*)wsalloc(base, &off, (size_t)NATOMS * FEAT * 4);
    float* v      = (float*)wsalloc(base, &off, (size_t)NATOMS * FEAT * 3 * 4);
    float* ds     = (float*)wsalloc(base, &off, (size_t)NATOMS * FEAT * 4);
    float* dv     = (float*)wsalloc(base, &off, (size_t)NATOMS * FEAT * 3 * 4);
    float* phi    = (float*)wsalloc(base, &off, (size_t)NATOMS * 3 * FEAT * 4); // also split
    float* tmp128 = (float*)wsalloc(base, &off, (size_t)NATOMS * FEAT * 4);
    float* mixb   = (float*)wsalloc(base, &off, (size_t)NATOMS * 2 * FEAT * 4); // also af/h2
    float* uv     = (float*)wsalloc(base, &off, (size_t)NATOMS * FEAT * 3 * 4);
    float* vv     = (float*)wsalloc(base, &off, (size_t)NATOMS * FEAT * 3 * 4);
    float* geo    = (float*)wsalloc(base, &off, (size_t)NEDGES * 24 * 4);

    _Float16* W1h  = (_Float16*)wsalloc(base, &off, (size_t)NCONV * FEAT * FEAT * 2);
    _Float16* W2h  = (_Float16*)wsalloc(base, &off, (size_t)NCONV * FEAT * 3 * FEAT * 2);
    _Float16* Uh   = (_Float16*)wsalloc(base, &off, (size_t)NCONV * FEAT * FEAT * 2);
    _Float16* Vh   = (_Float16*)wsalloc(base, &off, (size_t)NCONV * FEAT * FEAT * 2);
    _Float16* Ws1h = (_Float16*)wsalloc(base, &off, (size_t)NCONV * 2 * FEAT * FEAT * 2);
    _Float16* Ws2h = (_Float16*)wsalloc(base, &off, (size_t)NCONV * FEAT * 3 * FEAT * 2);
    _Float16* Wr1h = (_Float16*)wsalloc(base, &off, (size_t)FEAT * FEAT * 2);
    _Float16* Wr2h = (_Float16*)wsalloc(base, &off, (size_t)FEAT * AFEA * 2);
    _Float16* Wf1h = (_Float16*)wsalloc(base, &off, (size_t)AFEA * HFEA * 2);
    _Float16* Wf2h = (_Float16*)wsalloc(base, &off, (size_t)HFEA * HFEA * 2);
    _Float16* Wf3h = (_Float16*)wsalloc(base, &off, (size_t)HFEA * NOUT * 2);

    auto cvt = [&](const float* src, _Float16* dst, long n) {
        cvt_f16_kernel<<<dim3((unsigned)((n + 255) / 256)), dim3(256), 0, stream>>>(src, dst, n);
    };
    cvt(W1,  W1h,  (long)NCONV * FEAT * FEAT);
    cvt(W2,  W2h,  (long)NCONV * FEAT * 3 * FEAT);
    cvt(U,   Uh,   (long)NCONV * FEAT * FEAT);
    cvt(V,   Vh,   (long)NCONV * FEAT * FEAT);
    cvt(Ws1, Ws1h, (long)NCONV * 2 * FEAT * FEAT);
    cvt(Ws2, Ws2h, (long)NCONV * FEAT * 3 * FEAT);
    cvt(Wr1, Wr1h, (long)FEAT * FEAT);
    cvt(Wr2, Wr2h, (long)FEAT * AFEA);
    cvt(Wf1, Wf1h, (long)AFEA * HFEA);
    cvt(Wf2, Wf2h, (long)HFEA * HFEA);
    cvt(Wf3, Wf3h, (long)HFEA * NOUT);

    const int Mtiles = NATOMS / 16;   // 1250
    auto gemm = [&](const float* A, long ar, long ac, const _Float16* B, int N, int K,
                    const float* bias, float* C, long crow, long ccol, int act) {
        dim3 grid((unsigned)((N + 63) / 64), (unsigned)((Mtiles + 3) / 4));
        if (K == 64)
            gemm64x64_wmma<64><<<grid, dim3(128), 0, stream>>>(A, ar, ac, B, N, bias, C, crow, ccol, act, Mtiles);
        else if (K == 256)
            gemm64x64_wmma<256><<<grid, dim3(128), 0, stream>>>(A, ar, ac, B, N, bias, C, crow, ccol, act, Mtiles);
        else
            gemm64x64_wmma<128><<<grid, dim3(128), 0, stream>>>(A, ar, ac, B, N, bias, C, crow, ccol, act, Mtiles);
    };
    auto fill0 = [&](float* p, long n) {
        fill0_kernel<<<dim3((unsigned)((n + 255) / 256)), dim3(256), 0, stream>>>(p, n);
    };

    // ---- init: s = embed[z], v = 0, edge geometry -------------------------
    embed_gather_kernel<<<dim3((NATOMS * FEAT + 255) / 256), dim3(256), 0, stream>>>(embd, z, s);
    fill0(v, (long)NATOMS * FEAT * 3);
    edge_geo_kernel<<<dim3((NEDGES + 255) / 256), dim3(256), 0, stream>>>(xyz, nbrs, geo);

    // ---- conv layers ------------------------------------------------------
    for (int l = 0; l < NCONV; ++l) {
        // phi = silu(s@W1+b1) @ W2 + b2
        gemm(s, FEAT, 1, W1h + (size_t)l * FEAT * FEAT, FEAT, FEAT, b1 + l * FEAT,
             tmp128, FEAT, 1, 1);
        gemm(tmp128, FEAT, 1, W2h + (size_t)l * FEAT * 3 * FEAT, 3 * FEAT, FEAT,
             b2 + l * 3 * FEAT, phi, 3 * FEAT, 1, 0);

        // edge messages -> ds/dv
        fill0(ds, (long)NATOMS * FEAT);
        fill0(dv, (long)NATOMS * FEAT * 3);
        edge_msg_kernel<<<dim3(NEDGES), dim3(128), 0, stream>>>(
            geo, nbrs, phi, v, Wd + (size_t)l * NRBF * 3 * FEAT, bd + l * 3 * FEAT, ds, dv);
        addip_kernel<<<dim3((NATOMS * FEAT + 255) / 256), dim3(256), 0, stream>>>(s, ds, (long)NATOMS * FEAT);
        addip_kernel<<<dim3((NATOMS * FEAT * 3 + 255) / 256), dim3(256), 0, stream>>>(v, dv, (long)NATOMS * FEAT * 3);

        // u_v = v@U, v_v = v@V  (strided einsum GEMMs, 3 spatial dims each)
        for (int d = 0; d < 3; ++d) {
            gemm(v + d, FEAT * 3, 3, Uh + (size_t)l * FEAT * FEAT, FEAT, FEAT,
                 nullptr, uv + d, FEAT * 3, 3, 0);
            gemm(v + d, FEAT * 3, 3, Vh + (size_t)l * FEAT * FEAT, FEAT, FEAT,
                 nullptr, vv + d, FEAT * 3, 3, 0);
        }

        // mixing MLP
        mix_kernel<<<dim3((NATOMS * FEAT + 255) / 256), dim3(256), 0, stream>>>(s, vv, mixb);
        gemm(mixb, 2 * FEAT, 1, Ws1h + (size_t)l * 2 * FEAT * FEAT, FEAT, 2 * FEAT,
             bs1 + l * FEAT, tmp128, FEAT, 1, 1);
        gemm(tmp128, FEAT, 1, Ws2h + (size_t)l * FEAT * 3 * FEAT, 3 * FEAT, FEAT,
             bs2 + l * 3 * FEAT, phi /*split*/, 3 * FEAT, 1, 0);

        update_kernel<<<dim3((NATOMS * FEAT + 255) / 256), dim3(256), 0, stream>>>(s, v, uv, vv, phi);
    }

    // ---- readout ----------------------------------------------------------
    gemm(s, FEAT, 1, Wr1h, FEAT, FEAT, br1, tmp128, FEAT, 1, 1);          // silu(s@Wr1+br1)
    gemm(tmp128, FEAT, 1, Wr2h, AFEA, FEAT, br2, mixb, AFEA, 1, 0);       // atom_fea
    gemm(mixb, AFEA, 1, Wf1h, HFEA, AFEA, bf1, tmp128, HFEA, 1, 1);       // h1
    gemm(tmp128, HFEA, 1, Wf2h, HFEA, HFEA, bf2, mixb, HFEA, 1, 1);       // h2
    gemm(mixb, HFEA, 1, Wf3h, NOUT, HFEA, bf3, out, NOUT, 1, 0);          // output
}
